// Llama3_56392920597109
// MI455X (gfx1250) — matrix operand order repaired and checked by reading, not compiled, not run on hardware
//
#include <hip/hip_runtime.h>
#include <stdint.h>
#include <stddef.h>

// ---------------- model constants ----------------
#define BB   2
#define SS   1024
#define DD   768
#define HH   12
#define KVH  3          // H // G
#define HDIM 64
#define HID_ 2048
#define VV   32000
#define LLAY 8
#define MM   (BB*SS)    // 2048 token rows
#define EOS_ 2
#define EPSF 1.1920929e-07f
#define VSPLIT 16
#define VCOLS  (VV/VSPLIT)      // 2000
#define VCHUNK (VCOLS/16)       // 125

typedef __attribute__((ext_vector_type(16))) __bf16 v16bf;
typedef __attribute__((ext_vector_type(8)))  float  v8f;
typedef __attribute__((ext_vector_type(4)))  float  v4f;
typedef __attribute__((ext_vector_type(4)))  __bf16 v4bf;
typedef __attribute__((ext_vector_type(2)))  __bf16 v2bf;

union Frag { uint4 q[2]; v16bf v; };

// ---- native bf16 conversion helpers (hardware cvt, not SW emulation) ----
__device__ __forceinline__ unsigned pack2bf(float a, float b) {
#if __has_builtin(__builtin_amdgcn_cvt_pk_bf16_f32)
  v2bf p = __builtin_amdgcn_cvt_pk_bf16_f32(a, b);
  return __builtin_bit_cast(unsigned, p);
#else
  v2bf p; p[0] = (__bf16)a; p[1] = (__bf16)b;
  return __builtin_bit_cast(unsigned, p);
#endif
}
__device__ __forceinline__ unsigned short f2bf_raw(float f) {
  __bf16 b = (__bf16)f;
  return __builtin_bit_cast(unsigned short, b);
}
__device__ __forceinline__ uint2 pack4bf(v4f v) {
  v4bf b = __builtin_convertvector(v, v4bf);
  return __builtin_bit_cast(uint2, b);
}
__device__ __forceinline__ unsigned short bf_extract(uint2 v, int i) {
  unsigned w = (i < 2) ? v.x : v.y;
  return (unsigned short)(w >> (16 * (i & 1)));
}
__device__ __forceinline__ float f4c(const float4& v, int i) {
  return (i == 0) ? v.x : (i == 1) ? v.y : (i == 2) ? v.z : v.w;
}
__device__ __forceinline__ v8f wmma_bf16(v16bf a, v16bf b, v8f c) {
  return __builtin_amdgcn_wmma_f32_16x16x32_bf16(false, a, false, b, (short)0, c, false, false);
}

// ---------------- embedding gather (vectorized) ----------------
__global__ __launch_bounds__(256) void k_embed(const float* __restrict__ emb,
                                               const int* __restrict__ tgt,
                                               float* __restrict__ x) {
  int row = blockIdx.x;
  int tok = tgt[row];
  const float4* src = (const float4*)(emb + (size_t)tok * DD);
  float4* dst = (float4*)(x + (size_t)row * DD);
  for (int i = threadIdx.x; i < DD / 4; i += 256) dst[i] = src[i];
}

// ---------------- rmsnorm -> bf16 (vectorized) ----------------
__global__ __launch_bounds__(256) void k_rmsnorm(const float* __restrict__ x,
                                                 const float* __restrict__ w,
                                                 unsigned short* __restrict__ out) {
  __shared__ float red[256];
  int row = blockIdx.x;
  const float4* xr4 = (const float4*)(x + (size_t)row * DD);
  const float4* w4  = (const float4*)w;
  float s = 0.f;
  for (int i = threadIdx.x; i < DD / 4; i += 256) {
    float4 v = xr4[i];
    s += v.x * v.x + v.y * v.y + v.z * v.z + v.w * v.w;
  }
  red[threadIdx.x] = s; __syncthreads();
  for (int o = 128; o > 0; o >>= 1) {
    if (threadIdx.x < o) red[threadIdx.x] += red[threadIdx.x + o];
    __syncthreads();
  }
  float scale = rsqrtf(red[0] / (float)DD + EPSF);
  uint2* orow = (uint2*)(out + (size_t)row * DD);
  for (int i = threadIdx.x; i < DD / 4; i += 256) {
    float4 v = xr4[i], ww = w4[i];
    v4f t = { v.x * scale * ww.x, v.y * scale * ww.y,
              v.z * scale * ww.z, v.w * scale * ww.w };
    orow[i] = pack4bf(t);
  }
}

// ---------------- WMMA GEMM: C[MxN] = A(bf16,MxK) * W(f32->bf16,KxN) (+bias)(+residual)
// block: 128 threads = 4 waves; block tile 64x64; wave tile 32x32; k-step 32
// LDS: As[m][k] row-major; Bs[n][k] K-major so fragments are contiguous b128 pairs.
__global__ __launch_bounds__(128) void k_gemm(const unsigned short* __restrict__ A,
                                              const float* __restrict__ W,
                                              const float* __restrict__ bias,
                                              const float* __restrict__ residual,
                                              float* __restrict__ C,
                                              int M, int N, int K) {
  __shared__ unsigned short As[64][40];
  __shared__ unsigned short Bs[64][40];
  int tid  = threadIdx.x;
  int lane = tid & 31;
  int wave = tid >> 5;
  int wm = wave >> 1, wn = wave & 1;
  int bm = blockIdx.y * 64, bn = blockIdx.x * 64;
  v8f acc[2][2] = {};
  int arow = lane & 15;
  int koff = (lane >> 4) * 8;
  int kb   = (lane >> 4) * 8;
  int kq = (tid >> 4) * 4;     // k sub-quad 0..28 for B staging
  int nq = (tid & 15) * 4;     // n sub-quad 0..60

  for (int k0 = 0; k0 < K; k0 += 32) {
    // A staging: 8 bf16 per uint4, 2 per thread
#pragma unroll
    for (int rep = 0; rep < 2; ++rep) {
      int i = tid + rep * 128;
      int r = i >> 2, c = (i & 3) * 8;
      *(uint4*)&As[r][c] = *(const uint4*)&A[(size_t)(bm + r) * K + k0 + c];
    }
    // B staging: 4x4 f32 sub-block -> bf16, transposed into Bs[n][k]
    {
      const float* wp = &W[(size_t)(k0 + kq) * N + bn + nq];
      float4 w0 = *(const float4*)(wp);
      float4 w1 = *(const float4*)(wp + N);
      float4 w2 = *(const float4*)(wp + 2 * (size_t)N);
      float4 w3 = *(const float4*)(wp + 3 * (size_t)N);
#pragma unroll
      for (int i2 = 0; i2 < 4; ++i2) {
        uint2 pk;
        pk.x = pack2bf(f4c(w0, i2), f4c(w1, i2));
        pk.y = pack2bf(f4c(w2, i2), f4c(w3, i2));
        *(uint2*)&Bs[nq + i2][kq] = pk;
      }
    }
    if (k0 + 32 < K) __builtin_prefetch(&W[(size_t)(k0 + 32 + kq) * N + bn + nq], 0, 1);
    __syncthreads();

    Frag fb[2];
#pragma unroll
    for (int ni = 0; ni < 2; ++ni) {
      int coln = wn * 32 + ni * 16 + (lane & 15);
      fb[ni].q[0] = *(const uint4*)&Bs[coln][kb];
      fb[ni].q[1] = *(const uint4*)&Bs[coln][kb + 16];
    }
#pragma unroll
    for (int mi = 0; mi < 2; ++mi) {
      Frag fa;
      int r = wm * 32 + mi * 16 + arow;
      fa.q[0] = *(const uint4*)&As[r][koff];
      fa.q[1] = *(const uint4*)&As[r][koff + 16];
      acc[mi][0] = wmma_bf16(fa.v, fb[0].v, acc[mi][0]);
      acc[mi][1] = wmma_bf16(fa.v, fb[1].v, acc[mi][1]);
    }
    __syncthreads();
  }
#pragma unroll
  for (int mi = 0; mi < 2; ++mi)
#pragma unroll
    for (int ni = 0; ni < 2; ++ni) {
      int col = bn + wn * 32 + ni * 16 + (lane & 15);
#pragma unroll
      for (int r = 0; r < 8; ++r) {
        int row = bm + wm * 32 + mi * 16 + r + ((lane >> 4) * 8);
        if (row < M && col < N) {
          float v = acc[mi][ni][r];
          if (bias)     v += bias[col];
          if (residual) v += residual[(size_t)row * N + col];
          C[(size_t)row * N + col] = v;
        }
      }
    }
}

// ---------------- RoPE + convert to bf16 (layout B,S,nheads,HD) ----------------
__global__ __launch_bounds__(256) void k_rope(const float* __restrict__ in,
                                              unsigned short* __restrict__ out,
                                              int nheads, int total_pairs) {
  int idx = blockIdx.x * 256 + threadIdx.x;
  if (idx >= total_pairs) return;
  int p = idx & 31;
  size_t base = (size_t)(idx >> 5) * HDIM;
  int s = (int)(((idx >> 5) / nheads) % SS);
  float theta = __powf(10000.0f, -(float)p / 16.0f);  // 10000^(-2*(2p)/64)
  float fr = (float)s * theta;
  float sn, c;
  __sincosf(fr, &sn, &c);
  float2 xv = *(const float2*)(in + base + 2 * p);
  *(unsigned*)(out + base + 2 * p) = pack2bf(xv.x * c - xv.y * sn, xv.x * sn + xv.y * c);
}

__global__ __launch_bounds__(256) void k_cvt(const float* __restrict__ in,
                                             unsigned short* __restrict__ out, int n4) {
  int i = blockIdx.x * 256 + threadIdx.x;
  if (i < n4) {
    float4 v = ((const float4*)in)[i];
    v4f t = { v.x, v.y, v.z, v.w };
    ((uint2*)out)[i] = pack4bf(t);
  }
}

// ---------------- flash attention: 1 wave per 16 query rows ----------------
// q: (B,S,H,HD) bf16; k,v: (B,S,KVH,HD) bf16; out: (B,S,D) bf16
__global__ __launch_bounds__(32) void k_attn(const unsigned short* __restrict__ q,
                                             const unsigned short* __restrict__ kk,
                                             const unsigned short* __restrict__ vv,
                                             const int* __restrict__ amask,
                                             unsigned short* __restrict__ ob) {
  __shared__ float sS[16][33];
  __shared__ unsigned short sP[16][40];
  __shared__ unsigned short sV[64][40];   // sV[d][key]
  __shared__ float sRow[16];
  int blk = blockIdx.x;
  int qt = blk % (SS / 16);
  int h  = (blk / (SS / 16)) % HH;
  int b  = blk / ((SS / 16) * HH);
  int kvh = h % KVH;                      // jnp.tile -> h % KV
  int lane = threadIdx.x;
  int qbase = qt * 16;
  int arow = lane & 15;
  int koff = (lane >> 4) * 8;
  int kb   = (lane >> 4) * 8;

  // Q fragments (HD=64 -> two K-steps), contiguous b128 pairs from global
  Frag fq[2];
  {
    const unsigned short* qp = q + ((size_t)(b * SS + qbase + arow) * HH + h) * HDIM;
#pragma unroll
    for (int st = 0; st < 2; ++st) {
      fq[st].q[0] = *(const uint4*)&qp[st * 32 + koff];
      fq[st].q[1] = *(const uint4*)&qp[st * 32 + koff + 16];
    }
  }
  v8f o[4] = {};
  float m = -__builtin_inff(), l = 0.f;
  const float scale = 0.125f;             // 1/sqrt(64)
  const size_t kvstride = (size_t)KVH * HDIM;

  for (int kb0 = 0; kb0 < qbase + 16; kb0 += 32) {
    // ---- scores: 16x32 = 2 N-tiles x 2 K-steps ----
    v8f c2[2] = {};
#pragma unroll
    for (int nt = 0; nt < 2; ++nt) {
      int key = kb0 + nt * 16 + (lane & 15);
      const unsigned short* kp = kk + ((size_t)(b * SS + key) * KVH + kvh) * HDIM + kb;
#pragma unroll
      for (int st = 0; st < 2; ++st) {
        Frag fk;
        fk.q[0] = *(const uint4*)&kp[st * 32];
        fk.q[1] = *(const uint4*)&kp[st * 32 + 16];
        c2[nt] = wmma_bf16(fq[st].v, fk.v, c2[nt]);
      }
    }
    // ---- mask + spill scores to LDS ----
#pragma unroll
    for (int nt = 0; nt < 2; ++nt) {
      int col = nt * 16 + (lane & 15);
      int key = kb0 + col;
      bool am = (amask[b * SS + key] != 0);
#pragma unroll
      for (int r = 0; r < 8; ++r) {
        int row = r + ((lane >> 4) * 8);
        float sc = c2[nt][r] * scale;
        if (!(am && key <= qbase + row)) sc = -__builtin_inff();
        sS[row][col] = sc;
      }
    }
    // ---- stage V chunk (32 keys x 64 d) transposed into sV[d][key] ----
#pragma unroll
    for (int rep = 0; rep < 4; ++rep) {
      int sb = lane + rep * 32;
      int kq = (sb >> 4) * 4;
      int dq = (sb & 15) * 4;
      const unsigned short* vp = vv + ((size_t)(b * SS + kb0 + kq) * KVH + kvh) * HDIM + dq;
      uint2 a0 = *(const uint2*)(vp);
      uint2 a1 = *(const uint2*)(vp + kvstride);
      uint2 a2 = *(const uint2*)(vp + 2 * kvstride);
      uint2 a3 = *(const uint2*)(vp + 3 * kvstride);
#pragma unroll
      for (int i2 = 0; i2 < 4; ++i2) {
        uint2 pk;
        pk.x = (unsigned)bf_extract(a0, i2) | ((unsigned)bf_extract(a1, i2) << 16);
        pk.y = (unsigned)bf_extract(a2, i2) | ((unsigned)bf_extract(a3, i2) << 16);
        *(uint2*)&sV[dq + i2][kq] = pk;
      }
    }
    __syncthreads();
    // ---- online softmax; write P as bf16 ----
    {
      int row = lane & 15;
      float tm = -__builtin_inff();
      for (int cc = 0; cc < 32; ++cc) tm = fmaxf(tm, sS[row][cc]);
      float mnew = fmaxf(m, tm);
      float corr = (mnew == -__builtin_inff()) ? 1.0f : __expf(m - mnew);
      float sum = 0.f;
      for (int cc = 0; cc < 32; ++cc) {
        float v = sS[row][cc];
        float p = (v == -__builtin_inff()) ? 0.f : __expf(v - mnew);
        if (lane < 16) sP[row][cc] = f2bf_raw(p);
        sum += p;
      }
      l = l * corr + sum;
      m = mnew;
      if (lane < 16) sRow[row] = corr;
    }
    __syncthreads();
    // ---- rescale accumulators ----
#pragma unroll
    for (int nt = 0; nt < 4; ++nt)
#pragma unroll
      for (int r = 0; r < 8; ++r)
        o[nt][r] *= sRow[r + ((lane >> 4) * 8)];
    // ---- O += P * V (both fragments contiguous b128 pairs) ----
    Frag fp;
    fp.q[0] = *(const uint4*)&sP[arow][koff];
    fp.q[1] = *(const uint4*)&sP[arow][koff + 16];
#pragma unroll
    for (int nt = 0; nt < 4; ++nt) {
      int d = nt * 16 + (lane & 15);
      Frag fv;
      fv.q[0] = *(const uint4*)&sV[d][kb];
      fv.q[1] = *(const uint4*)&sV[d][kb + 16];
      o[nt] = wmma_bf16(fp.v, fv.v, o[nt]);
    }
    __syncthreads();
  }
  if (lane < 16) sRow[lane] = (l > 0.f) ? (1.0f / l) : 0.f;
  __syncthreads();
#pragma unroll
  for (int nt = 0; nt < 4; ++nt) {
    int d = h * HDIM + nt * 16 + (lane & 15);
#pragma unroll
    for (int r = 0; r < 8; ++r) {
      int row = r + ((lane >> 4) * 8);
      ob[(size_t)(b * SS + qbase + row) * DD + d] = f2bf_raw(o[nt][r] * sRow[row]);
    }
  }
}

// ---------------- silu(g) * u -> bf16 (vectorized) ----------------
__global__ __launch_bounds__(256) void k_silugate(const float* __restrict__ g,
                                                  const float* __restrict__ u,
                                                  unsigned short* __restrict__ out, int n4) {
  int i = blockIdx.x * 256 + threadIdx.x;
  if (i < n4) {
    float4 gv = ((const float4*)g)[i];
    float4 uv = ((const float4*)u)[i];
    v4f t = { gv.x / (1.0f + __expf(-gv.x)) * uv.x,
              gv.y / (1.0f + __expf(-gv.y)) * uv.y,
              gv.z / (1.0f + __expf(-gv.z)) * uv.z,
              gv.w / (1.0f + __expf(-gv.w)) * uv.w };
    ((uint2*)out)[i] = pack4bf(t);
  }
}

// ---------------- streaming LM head + logsumexp partials ----------------
// grid (MM/32, VSPLIT), 1 wave; 32 rows per block; weights staged transposed in LDS
__global__ __launch_bounds__(32) void k_lmhead(const unsigned short* __restrict__ xn,
                                               const float* __restrict__ lmw,
                                               const float* __restrict__ lmb,
                                               const int* __restrict__ labels,
                                               float* __restrict__ pm,
                                               float* __restrict__ pl,
                                               float* __restrict__ plab) {
  __shared__ unsigned short sA[32][DD + 16];
  __shared__ unsigned short sW[16][40];    // sW[n][k]
  __shared__ float sT[32][17];
  int lane = threadIdx.x;
  int rowbase = blockIdx.x * 32;
  int split = blockIdx.y;
  int col0 = split * VCOLS;
  int arow = lane & 15;
  int koff = (lane >> 4) * 8;
  int kb   = (lane >> 4) * 8;
  int wkq = (lane >> 2) * 4;   // 0..28 (k quad for weight staging)
  int wnq = (lane & 3) * 4;    // 0..12 (n quad)

  // stage activations: 32 x 768 bf16 via uint4
  for (int i = lane; i < 32 * (DD / 8); i += 32) {
    int r = i / (DD / 8), c = (i % (DD / 8)) * 8;
    *(uint4*)&sA[r][c] = *(const uint4*)&xn[(size_t)(rowbase + r) * DD + c];
  }
  __syncthreads();

  // shifted label for this lane's row
  int gro = rowbase + lane;
  int bidx = gro / SS, sidx = gro % SS;
  int lab = (sidx < SS - 1) ? labels[bidx * SS + sidx + 1] : EOS_;

  float m = -__builtin_inff(), l = 0.f, lablog = -__builtin_inff();

  for (int cc = 0; cc < VCHUNK; ++cc) {
    int colc = col0 + cc * 16;
    v8f acc[2] = {};
    for (int ks = 0; ks < DD / 32; ++ks) {
      // stage 32(k) x 16(n) weight tile transposed as bf16
      {
        const float* wp = &lmw[(size_t)(ks * 32 + wkq) * VV + colc + wnq];
        float4 w0 = *(const float4*)(wp);
        float4 w1 = *(const float4*)(wp + VV);
        float4 w2 = *(const float4*)(wp + 2 * (size_t)VV);
        float4 w3 = *(const float4*)(wp + 3 * (size_t)VV);
#pragma unroll
        for (int i2 = 0; i2 < 4; ++i2) {
          uint2 pk;
          pk.x = pack2bf(f4c(w0, i2), f4c(w1, i2));
          pk.y = pack2bf(f4c(w2, i2), f4c(w3, i2));
          *(uint2*)&sW[wnq + i2][wkq] = pk;
        }
      }
      __syncthreads();
      Frag fb;
      int coln = lane & 15;
      fb.q[0] = *(const uint4*)&sW[coln][kb];
      fb.q[1] = *(const uint4*)&sW[coln][kb + 16];
#pragma unroll
      for (int mi = 0; mi < 2; ++mi) {
        Frag fa;
        int r = mi * 16 + arow;
        fa.q[0] = *(const uint4*)&sA[r][ks * 32 + koff];
        fa.q[1] = *(const uint4*)&sA[r][ks * 32 + koff + 16];
        acc[mi] = wmma_bf16(fa.v, fb.v, acc[mi]);
      }
      __syncthreads();
    }
    // spill logits (+bias) to LDS
    {
      int col = colc + (lane & 15);
      float bias = lmb[col];
#pragma unroll
      for (int mi = 0; mi < 2; ++mi)
#pragma unroll
        for (int r = 0; r < 8; ++r)
          sT[mi * 16 + r + ((lane >> 4) * 8)][lane & 15] = acc[mi][r] + bias;
    }
    __syncthreads();
    // per-row online logsumexp (lane owns row=lane)
    {
      float tm = -__builtin_inff();
      for (int c16 = 0; c16 < 16; ++c16) tm = fmaxf(tm, sT[lane][c16]);
      float mnew = fmaxf(m, tm);
      float corr = __expf(m - mnew);
      float sum = 0.f;
      for (int c16 = 0; c16 < 16; ++c16) sum += __expf(sT[lane][c16] - mnew);
      l = l * corr + sum;
      m = mnew;
      if (lab >= colc && lab < colc + 16) lablog = sT[lane][lab - colc];
    }
    __syncthreads();
  }
  pm[(size_t)gro * VSPLIT + split]   = m;
  pl[(size_t)gro * VSPLIT + split]   = l;
  plab[(size_t)gro * VSPLIT + split] = lablog;
}

// ---------------- combine partials -> mean NLL ----------------
__global__ __launch_bounds__(256) void k_finalize(const float* __restrict__ pm,
                                                  const float* __restrict__ pl,
                                                  const float* __restrict__ plab,
                                                  float* __restrict__ out) {
  __shared__ float red[256];
  float acc = 0.f;
  for (int row = threadIdx.x; row < MM; row += 256) {
    float M = -__builtin_inff();
    for (int s = 0; s < VSPLIT; ++s) M = fmaxf(M, pm[(size_t)row * VSPLIT + s]);
    float L = 0.f, lab = -__builtin_inff();
    for (int s = 0; s < VSPLIT; ++s) {
      L += pl[(size_t)row * VSPLIT + s] * __expf(pm[(size_t)row * VSPLIT + s] - M);
      lab = fmaxf(lab, plab[(size_t)row * VSPLIT + s]);
    }
    acc += (M + __logf(L)) - lab;
  }
  red[threadIdx.x] = acc; __syncthreads();
  for (int o = 128; o > 0; o >>= 1) {
    if (threadIdx.x < o) red[threadIdx.x] += red[threadIdx.x + o];
    __syncthreads();
  }
  if (threadIdx.x == 0) out[0] = red[0] / (float)MM;
}

// ---------------- host orchestration ----------------
extern "C" void kernel_launch(void* const* d_in, const int* in_sizes, int n_in,
                              void* d_out, int out_size, void* d_ws, size_t ws_size,
                              hipStream_t stream) {
  (void)in_sizes; (void)n_in; (void)out_size; (void)ws_size;
  const float* emb  = (const float*)d_in[0];
  const float* wq   = (const float*)d_in[1];
  const float* bq   = (const float*)d_in[2];
  const float* wk   = (const float*)d_in[3];
  const float* bk   = (const float*)d_in[4];
  const float* wv   = (const float*)d_in[5];
  const float* bv   = (const float*)d_in[6];
  const float* wo   = (const float*)d_in[7];
  const float* bo   = (const float*)d_in[8];
  const float* n1   = (const float*)d_in[9];
  const float* n2   = (const float*)d_in[10];
  const float* w1   = (const float*)d_in[11];
  const float* vw   = (const float*)d_in[12];
  const float* w2   = (const float*)d_in[13];
  const float* normw= (const float*)d_in[14];
  const float* lmw  = (const float*)d_in[15];
  const float* lmb  = (const float*)d_in[16];
  const int*   tgt  = (const int*)d_in[17];
  const int*   amsk = (const int*)d_in[18];
  const int*   labs = (const int*)d_in[19];
  float* out = (float*)d_out;

  char* wsp = (char*)d_ws;
  size_t off = 0;
  auto alloc = [&](size_t bytes) -> void* {
    void* p = wsp + off;
    off += (bytes + 255) & ~(size_t)255;
    return p;
  };
  float*          x   = (float*)alloc((size_t)MM * DD * 4);
  unsigned short* xn  = (unsigned short*)alloc((size_t)MM * DD * 2);
  float*          qf  = (float*)alloc((size_t)MM * DD * 4);
  float*          kf  = (float*)alloc((size_t)MM * KVH * HDIM * 4);
  float*          vf  = (float*)alloc((size_t)MM * KVH * HDIM * 4);
  unsigned short* qb  = (unsigned short*)alloc((size_t)MM * DD * 2);
  unsigned short* kb_ = (unsigned short*)alloc((size_t)MM * KVH * HDIM * 2);
  unsigned short* vb_ = (unsigned short*)alloc((size_t)MM * KVH * HDIM * 2);
  unsigned short* ob  = (unsigned short*)alloc((size_t)MM * DD * 2);
  float*          g   = (float*)alloc((size_t)MM * HID_ * 4);
  float*          u   = (float*)alloc((size_t)MM * HID_ * 4);
  unsigned short* tb  = (unsigned short*)alloc((size_t)MM * HID_ * 2);
  float*          pm  = (float*)alloc((size_t)MM * VSPLIT * 4);
  float*          pl  = (float*)alloc((size_t)MM * VSPLIT * 4);
  float*          plb = (float*)alloc((size_t)MM * VSPLIT * 4);

  k_embed<<<MM, 256, 0, stream>>>(emb, tgt, x);

  for (int layer = 0; layer < LLAY; ++layer) {
    const float* Wq = wq + (size_t)layer * DD * DD;
    const float* Bq = bq + (size_t)layer * DD;
    const float* Wk = wk + (size_t)layer * DD * (KVH * HDIM);
    const float* Bk = bk + (size_t)layer * (KVH * HDIM);
    const float* Wv = wv + (size_t)layer * DD * (KVH * HDIM);
    const float* Bv = bv + (size_t)layer * (KVH * HDIM);
    const float* Wo = wo + (size_t)layer * DD * DD;
    const float* Bo = bo + (size_t)layer * DD;
    const float* N1 = n1 + (size_t)layer * DD;
    const float* N2 = n2 + (size_t)layer * DD;
    const float* W1 = w1 + (size_t)layer * DD * HID_;
    const float* Vw = vw + (size_t)layer * DD * HID_;
    const float* W2 = w2 + (size_t)layer * HID_ * DD;

    k_rmsnorm<<<MM, 256, 0, stream>>>(x, N1, xn);

    k_gemm<<<dim3(DD / 64, MM / 64), 128, 0, stream>>>(xn, Wq, Bq, nullptr, qf, MM, DD, DD);
    k_gemm<<<dim3((KVH * HDIM) / 64, MM / 64), 128, 0, stream>>>(xn, Wk, Bk, nullptr, kf, MM, KVH * HDIM, DD);
    k_gemm<<<dim3((KVH * HDIM) / 64, MM / 64), 128, 0, stream>>>(xn, Wv, Bv, nullptr, vf, MM, KVH * HDIM, DD);

    k_rope<<<(MM * HH * 32 + 255) / 256, 256, 0, stream>>>(qf, qb, HH, MM * HH * 32);
    k_rope<<<(MM * KVH * 32 + 255) / 256, 256, 0, stream>>>(kf, kb_, KVH, MM * KVH * 32);
    k_cvt<<<(MM * KVH * HDIM / 4 + 255) / 256, 256, 0, stream>>>(vf, vb_, MM * KVH * HDIM / 4);

    k_attn<<<BB * HH * (SS / 16), 32, 0, stream>>>(qb, kb_, vb_, amsk, ob);

    k_gemm<<<dim3(DD / 64, MM / 64), 128, 0, stream>>>(ob, Wo, Bo, x, x, MM, DD, DD);

    k_rmsnorm<<<MM, 256, 0, stream>>>(x, N2, xn);
    k_gemm<<<dim3(HID_ / 64, MM / 64), 128, 0, stream>>>(xn, W1, nullptr, nullptr, g, MM, HID_, DD);
    k_gemm<<<dim3(HID_ / 64, MM / 64), 128, 0, stream>>>(xn, Vw, nullptr, nullptr, u, MM, HID_, DD);
    k_silugate<<<(MM * HID_ / 4 + 255) / 256, 256, 0, stream>>>(g, u, tb, MM * HID_ / 4);
    k_gemm<<<dim3(DD / 64, MM / 64), 128, 0, stream>>>(tb, W2, nullptr, x, x, MM, DD, HID_);
  }

  k_rmsnorm<<<MM, 256, 0, stream>>>(x, normw, xn);
  k_lmhead<<<dim3(MM / 32, VSPLIT), 32, 0, stream>>>(xn, lmw, lmb, labs, pm, pl, plb);
  k_finalize<<<1, 256, 0, stream>>>(pm, pl, plb, out);
}
